// MultiHeadedRelAttention_72816875537089
// MI455X (gfx1250) — matrix operand order repaired and checked by reading, not compiled, not run on hardware
//
#include <hip/hip_runtime.h>
#include <hip/hip_bf16.h>
#include <stdint.h>

// MultiHeadedAttention forward for MI455X (gfx1250), bf16 WMMA path.
// B=1, S=2048, D=1024, H=16, dh=64. Outputs: out (S*D f32) then attn (H*S*S f32).

#define SEQ    2048
#define NHEAD  16
#define DHEAD  64
#define DMODEL 1024
#define NEGINF -1e18f

typedef __attribute__((ext_vector_type(16))) __bf16 v16bf;
typedef __attribute__((ext_vector_type(8)))  float  v8f;
typedef __hip_bfloat16 bf16_t;

// ---------------------------------------------------------------------------
// Fragment loaders for V_WMMA_F32_16X16X32_BF16 (wave32).
// A-matrix 16x32 (MxK): lanes 0-15 = rows M=0..15 holding
// K=0..15 contiguous; lanes 16-31 same rows holding K=16..31.
// B-matrix 32x16 (KxN), same k order as A: lane n (0-15) =
// column N=n holding K=0..15 contiguous; lanes 16-31 hold K=16..31.
// Both loaders read *rows of length K* (NT GEMM: D[m,n]=sum_k A[m,k]*B[n,k]).
// ---------------------------------------------------------------------------
__device__ __forceinline__ v16bf load_fragA(const bf16_t* base, int row0, int ld, int k0) {
  int lane = threadIdx.x & 31;
  int l = lane & 15, hi = lane >> 4;
  const uint32_t* q = (const uint32_t*)(base + (size_t)(row0 + l) * ld + k0 + hi * 16);
  union { uint32_t u[8]; v16bf v; } f;
#pragma unroll
  for (int i = 0; i < 8; ++i) f.u[i] = q[i];
  return f.v;
}

__device__ __forceinline__ v16bf load_fragB(const bf16_t* base, int row0, int ld, int k0) {
  int lane = threadIdx.x & 31;
  int l = lane & 15, hi = lane >> 4;
  const uint32_t* q = (const uint32_t*)(base + (size_t)(row0 + l) * ld + k0 + hi * 16);
  union { uint32_t u[8]; v16bf v; } f;
#pragma unroll
  for (int i = 0; i < 8; ++i) f.u[i] = q[i];
  return f.v;
}

__device__ __forceinline__ v8f wmma_bf16(v16bf a, v16bf b, v8f c) {
  // (neg_a, A, neg_b, B, c_mod, C, reuse_a, reuse_b)
  return __builtin_amdgcn_wmma_f32_16x16x32_bf16(false, a, false, b, (short)0, c, false, false);
}

// ---------------------------------------------------------------------------
// f32 -> bf16 cast
// ---------------------------------------------------------------------------
__global__ void cast_kernel(const float* __restrict__ in, bf16_t* __restrict__ out, int n) {
  int i = blockIdx.x * blockDim.x + threadIdx.x;
  if (i < n) out[i] = __float2bfloat16(in[i]);
}

// ---------------------------------------------------------------------------
// Register-blocked NT GEMM: C = (A(MxK,row) * B(NxK,row)^T + bias[n]) * scale
//   - 64x64 output tile per wave: 4 A-frags x 4 B-frags -> 16 WMMAs per
//     8 fragment fetches (32 flop/B fetched, 4x the naive 1-frag tile).
//   - 4 waves / block arranged 2x2 -> 128x128 block tile for L0/L2 locality.
//   - global_prefetch_b8 for the next K slice.
// mode 0: store bf16 row-major   C[m*ldc+n]
// mode 1: store bf16 transposed  C[n*ldc+m]   (used to build V^T)
// mode 2: store f32  row-major   C[m*ldc+n]   (final output)
// M,N multiples of 128; K multiple of 32.
// ---------------------------------------------------------------------------
__global__ __launch_bounds__(128) void gemm_nt_kernel(
    const bf16_t* __restrict__ A, const bf16_t* __restrict__ B,
    const float* __restrict__ bias, void* __restrict__ C,
    int M, int N, int K, int lda, int ldb, int ldc, float scale, int mode)
{
  int wid  = threadIdx.x >> 5;
  int lane = threadIdx.x & 31;
  int l = lane & 15, hi = lane >> 4;

  int tilesN = N >> 7;                       // 128-wide block tiles
  int bm = blockIdx.x / tilesN, bn = blockIdx.x % tilesN;
  int m0 = bm * 128 + (wid >> 1) * 64;       // this wave's 64x64 tile
  int n0 = bn * 128 + (wid & 1) * 64;

  v8f acc[4][4] = {};
#pragma unroll 1
  for (int k0 = 0; k0 < K; k0 += 32) {
    if (k0 + 32 < K) {                       // prefetch next K slice (global_prefetch_b8)
      __builtin_prefetch(A + (size_t)(m0 + (lane & 63)) * lda + k0 + 32, 0, 3);
      __builtin_prefetch(B + (size_t)(n0 + (lane & 63)) * ldb + k0 + 32, 0, 3);
    }
    v16bf a[4], b[4];
#pragma unroll
    for (int i = 0; i < 4; ++i) a[i] = load_fragA(A, m0 + 16 * i, lda, k0);
#pragma unroll
    for (int j = 0; j < 4; ++j) b[j] = load_fragB(B, n0 + 16 * j, ldb, k0);
#pragma unroll
    for (int i = 0; i < 4; ++i)
#pragma unroll
      for (int j = 0; j < 4; ++j)
        acc[i][j] = wmma_bf16(a[i], b[j], acc[i][j]);
  }

#pragma unroll
  for (int j = 0; j < 4; ++j) {
    int n = n0 + 16 * j + l;
    float bv = bias ? bias[n] : 0.0f;
#pragma unroll
    for (int i = 0; i < 4; ++i) {
#pragma unroll
      for (int r = 0; r < 8; ++r) {
        int m = m0 + 16 * i + r + hi * 8;    // C/D layout: lane n = lane&15, M = r (+8 hi half)
        float v = (acc[i][j][r] + bv) * scale;
        if (mode == 0)      ((bf16_t*)C)[(size_t)m * ldc + n] = __float2bfloat16(v);
        else if (mode == 1) ((bf16_t*)C)[(size_t)n * ldc + m] = __float2bfloat16(v);
        else                ((float*)C)[(size_t)m * ldc + n] = v;
      }
    }
  }
}

// ---------------------------------------------------------------------------
// Attention core: one block = (head h, 16-query tile q0).
// Waves 0..3 each produce 32 of the 128 16x16 score tiles -> LDS (f32),
// mask applied inline. Barrier softmax over the 2048-wide rows, attn written
// to global (f32), P cast to bf16 in LDS, then each wave computes one 16x16
// ctx fragment (64 WMMAs against V^T).
// LDS: 16*2048*4 + 16*2048*2 + 512 = ~197 KB (CDNA5 allows 320 KB/WG).
// ---------------------------------------------------------------------------
__global__ __launch_bounds__(128) void attn_kernel(
    const bf16_t* __restrict__ Qb, const bf16_t* __restrict__ Kb,
    const bf16_t* __restrict__ Vtb, const unsigned char* __restrict__ mask,
    float* __restrict__ attn_out, bf16_t* __restrict__ ctxb)
{
  __shared__ float  sc[16 * SEQ];     // 128 KB score strip
  __shared__ bf16_t pb[16 * SEQ];     //  64 KB softmaxed probs (bf16)
  __shared__ float  red[16 * 8];

  int h  = blockIdx.x >> 7;
  int q0 = (blockIdx.x & 127) << 4;
  int tid = threadIdx.x;
  int wid = tid >> 5, lane = tid & 31;
  int l = lane & 15, hi = lane >> 4;

  // Q fragments for this 16-row tile (head h occupies columns h*64..h*64+63)
  v16bf aq0 = load_fragA(Qb + h * DHEAD, q0, DMODEL, 0);
  v16bf aq1 = load_fragA(Qb + h * DHEAD, q0, DMODEL, 32);

  // ---- scores = Q K^T (Q pre-scaled), mask, -> LDS ----
  for (int nt = wid * 32; nt < wid * 32 + 32; ++nt) {
    int n0 = nt << 4;
    v16bf b0 = load_fragB(Kb + h * DHEAD, n0, DMODEL, 0);
    v16bf b1 = load_fragB(Kb + h * DHEAD, n0, DMODEL, 32);
    v8f acc = {};
    acc = wmma_bf16(aq0, b0, acc);
    acc = wmma_bf16(aq1, b1, acc);
    int n = n0 + l;
#pragma unroll
    for (int r = 0; r < 8; ++r) {
      int m = r + hi * 8;
      bool mk = mask[(size_t)(q0 + m) * SEQ + n] != 0;   // jnp bool = 1 byte
      sc[m * SEQ + n] = mk ? NEGINF : acc[r];
    }
  }
  __syncthreads();

  // ---- softmax: 8 threads per row ----
  int row = tid >> 3, g = tid & 7;
  float mx = -3.4e38f;
  for (int c = g; c < SEQ; c += 8) mx = fmaxf(mx, sc[row * SEQ + c]);
  red[row * 8 + g] = mx;
  __syncthreads();
  mx = red[row * 8];
#pragma unroll
  for (int i = 1; i < 8; ++i) mx = fmaxf(mx, red[row * 8 + i]);
  __syncthreads();                       // all reads of red done before reuse

  float sum = 0.f;
  for (int c = g; c < SEQ; c += 8) {
    float e = __expf(sc[row * SEQ + c] - mx);
    sc[row * SEQ + c] = e;
    sum += e;
  }
  red[row * 8 + g] = sum;
  __syncthreads();
  sum = 0.f;
#pragma unroll
  for (int i = 0; i < 8; ++i) sum += red[row * 8 + i];
  float inv = 1.0f / sum;

  float* arow = attn_out + ((size_t)h * SEQ + q0 + row) * SEQ;
  for (int c = g; c < SEQ; c += 8) {
    float p = sc[row * SEQ + c] * inv;
    arow[c] = p;                          // materialize attn (required output)
    pb[row * SEQ + c] = __float2bfloat16(p);
  }
  __syncthreads();

  // ---- ctx = P * V  (NT vs V^T rows): each wave one 16x16 fragment ----
  int nd = wid << 4;                      // column offset within dh=64
  v8f cacc = {};
  for (int kt = 0; kt < SEQ / 32; ++kt) {
    v16bf a = load_fragA(pb, 0, SEQ, kt * 32);                  // P tile from LDS
    v16bf b = load_fragB(Vtb, h * DHEAD + nd, SEQ, kt * 32);    // V^T rows = d
    cacc = wmma_bf16(a, b, cacc);
  }
#pragma unroll
  for (int r = 0; r < 8; ++r) {
    int m = q0 + r + hi * 8;
    ctxb[(size_t)m * DMODEL + h * DHEAD + nd + l] = __float2bfloat16(cacc[r]);
  }
}

// ---------------------------------------------------------------------------
// Launcher. d_in order: inputs, mask, Wq, bq, Wk, bk, Wv, bv, Wo, bo
// d_out: out (2048*1024 f32) then attn (16*2048*2048 f32).
// Workspace layout (28 MB used):
//   [0,4M)   Xb    bf16 2048x1024
//   [4,6M)   Wqb   [6,8M) Wkb  [8,10M) Wvb  [10,12M) Wob   (bf16 1024x1024)
//   [12,16M) Qb (scaled)  [16,20M) Kb  [20,24M) Vtb (d x S)  [24,28M) ctx
// ---------------------------------------------------------------------------
extern "C" void kernel_launch(void* const* d_in, const int* in_sizes, int n_in,
                              void* d_out, int out_size, void* d_ws, size_t ws_size,
                              hipStream_t stream) {
  (void)in_sizes; (void)n_in; (void)out_size; (void)ws_size;
  const float* X  = (const float*)d_in[0];
  const unsigned char* mask = (const unsigned char*)d_in[1];
  const float* Wq = (const float*)d_in[2];
  const float* bq = (const float*)d_in[3];
  const float* Wk = (const float*)d_in[4];
  const float* bk = (const float*)d_in[5];
  const float* Wv = (const float*)d_in[6];
  const float* bv = (const float*)d_in[7];
  const float* Wo = (const float*)d_in[8];
  const float* bo = (const float*)d_in[9];

  char* ws = (char*)d_ws;
  bf16_t* Xb  = (bf16_t*)(ws);
  bf16_t* Wqb = (bf16_t*)(ws + ((size_t)4  << 20));
  bf16_t* Wkb = (bf16_t*)(ws + ((size_t)6  << 20));
  bf16_t* Wvb = (bf16_t*)(ws + ((size_t)8  << 20));
  bf16_t* Wob = (bf16_t*)(ws + ((size_t)10 << 20));
  bf16_t* Qb  = (bf16_t*)(ws + ((size_t)12 << 20));
  bf16_t* Kb  = (bf16_t*)(ws + ((size_t)16 << 20));
  bf16_t* Vtb = (bf16_t*)(ws + ((size_t)20 << 20));
  bf16_t* Cxb = (bf16_t*)(ws + ((size_t)24 << 20));

  float* out  = (float*)d_out;
  float* attn = out + (size_t)SEQ * DMODEL;

  int nX = SEQ * DMODEL, nW = DMODEL * DMODEL;
  cast_kernel<<<(nX + 255) / 256, 256, 0, stream>>>(X,  Xb,  nX);
  cast_kernel<<<(nW + 255) / 256, 256, 0, stream>>>(Wq, Wqb, nW);
  cast_kernel<<<(nW + 255) / 256, 256, 0, stream>>>(Wk, Wkb, nW);
  cast_kernel<<<(nW + 255) / 256, 256, 0, stream>>>(Wv, Wvb, nW);
  cast_kernel<<<(nW + 255) / 256, 256, 0, stream>>>(Wo, Wob, nW);

  int gemmBlocks = (SEQ / 128) * (DMODEL / 128);  // 128 blocks, 4 waves each
  // Q = (X Wq^T + bq) * 1/sqrt(64)
  gemm_nt_kernel<<<gemmBlocks, 128, 0, stream>>>(Xb, Wqb, bq, (void*)Qb,
      SEQ, DMODEL, DMODEL, DMODEL, DMODEL, DMODEL, 0.125f, 0);
  gemm_nt_kernel<<<gemmBlocks, 128, 0, stream>>>(Xb, Wkb, bk, (void*)Kb,
      SEQ, DMODEL, DMODEL, DMODEL, DMODEL, DMODEL, 1.0f, 0);
  // V stored transposed: Vt[d][s]
  gemm_nt_kernel<<<gemmBlocks, 128, 0, stream>>>(Xb, Wvb, bv, (void*)Vtb,
      SEQ, DMODEL, DMODEL, DMODEL, DMODEL, SEQ, 1.0f, 1);

  attn_kernel<<<NHEAD * (SEQ / 16), 128, 0, stream>>>(Qb, Kb, Vtb, mask, attn, Cxb);

  // out = ctx Wo^T + bo  (f32)
  gemm_nt_kernel<<<gemmBlocks, 128, 0, stream>>>(Cxb, Wob, bo, (void*)out,
      SEQ, DMODEL, DMODEL, DMODEL, DMODEL, DMODEL, 1.0f, 2);
}
